// PointNetSetAbstractionMsgWithLGE_68676527063208
// MI455X (gfx1250) — matrix operand order repaired and checked by reading, not compiled, not run on hardware
//
#include <hip/hip_runtime.h>
#include <hip/hip_bf16.h>

typedef __attribute__((ext_vector_type(16))) _Float16 v16h;
typedef __attribute__((ext_vector_type(8)))  float    v8f;
typedef unsigned int uint;

#define BB   8
#define NN   4096
#define SS   1024
#define CIN  64
#define OUT_XYZ_ELEMS (BB * 3 * SS)     // 24576
#define TOTAL_OUT_CH  320

// ---------------------------------------------------------------------------
// 1. Gather new_xyz / new_pts at fps_index; also write transposed new_xyz out.
// ---------------------------------------------------------------------------
__global__ void gather_new_kernel(const float* __restrict__ xyz,   // [B,3,N]
                                  const float* __restrict__ pts,   // [B,CIN,N]
                                  const int*   __restrict__ fps,   // [B,S]
                                  float* __restrict__ newxyz,      // [B*S,3]
                                  float* __restrict__ newpts,      // [B*S,CIN]
                                  float* __restrict__ out)         // d_out
{
    int t = blockIdx.x * 256 + threadIdx.x;          // one thread per (b,s)
    if (t >= BB * SS) return;
    int b = t >> 10;          // S = 1024
    int s = t & 1023;
    int idx = fps[t];
    #pragma unroll
    for (int c = 0; c < 3; ++c) {
        float v = xyz[((long)b * 3 + c) * NN + idx];
        newxyz[(long)t * 3 + c] = v;
        out[((long)b * 3 + c) * SS + s] = v;         // new_xyz transposed [B,3,S]
    }
    for (int c = 0; c < CIN; ++c)
        newpts[(long)t * CIN + c] = pts[((long)b * CIN + c) * NN + idx];
}

// ---------------------------------------------------------------------------
// 2. Ball query: one wave32 per query; keeps first K in-radius indices in
//    increasing index order (matches reference sort-then-truncate semantics).
// ---------------------------------------------------------------------------
__global__ void ball_query_kernel(const float* __restrict__ xyz,     // [B,3,N]
                                  const float* __restrict__ newxyz,  // [B*S,3]
                                  int* __restrict__ gidx,            // [B*S,K]
                                  float r2, int K)
{
    int lane = threadIdx.x & 31;
    int wave = threadIdx.x >> 5;
    int q = blockIdx.x * 8 + wave;                   // query id in [0, B*S)
    if (q >= BB * SS) return;
    int b = q >> 10;
    float qx = newxyz[(long)q * 3 + 0];
    float qy = newxyz[(long)q * 3 + 1];
    float qz = newxyz[(long)q * 3 + 2];
    const float* px = xyz + (long)b * 3 * NN;
    const float* py = px + NN;
    const float* pz = py + NN;
    long base = (long)q * K;

    int cnt = 0;
    int firstIdx = 0x7fffffff;
    for (int n0 = 0; n0 < NN && cnt < K; n0 += 32) {
        int n = n0 + lane;
        float dx = qx - px[n];
        float dy = qy - py[n];
        float dz = qz - pz[n];
        float d2 = dx * dx + dy * dy + dz * dz;
        bool in = (d2 <= r2);
        unsigned mask = (unsigned)__ballot((int)in);
        int pre = __popc(mask & ((1u << lane) - 1u));
        int pos = cnt + pre;
        if (in && pos < K) gidx[base + pos] = n;
        if (in && pos == 0) firstIdx = n;
        cnt += __popc(mask);
    }
    // wave-min reduce for first valid index
    for (int off = 16; off > 0; off >>= 1) {
        int o = __shfl_xor(firstIdx, off, 32);
        firstIdx = (o < firstIdx) ? o : firstIdx;
    }
    if (cnt == 0) firstIdx = 0;
    for (int j = cnt + lane; j < K; j += 32) gidx[base + j] = firstIdx;
}

// ---------------------------------------------------------------------------
// 3. Build feature rows [M x 160] f16:
//    [0..63]   sigmoid(diff)   * (pts[idx] - center)
//    [64..127] sigmoid(center) * center
//    [128..130] (xyz[idx]-new_xyz)/radius ; [131..159] = 0
// ---------------------------------------------------------------------------
__global__ void feat_build_kernel(const float* __restrict__ xyz,
                                  const float* __restrict__ pts,
                                  const float* __restrict__ newxyz,
                                  const float* __restrict__ newpts,
                                  const int*   __restrict__ gidx,
                                  const float* __restrict__ dlogit,
                                  const float* __restrict__ clogit,
                                  float inv_radius, int K,
                                  _Float16* __restrict__ A)          // ld = 160
{
    long row = (long)blockIdx.x * 4 + threadIdx.y;   // M divisible by 4
    int  c   = threadIdx.x;                          // 0..63
    int  bs  = (int)(row / K);
    int  b   = bs >> 10;
    int  idx = gidx[row];
    float da = 1.f / (1.f + __expf(-dlogit[0]));
    float ca = 1.f / (1.f + __expf(-clogit[0]));
    float pv = pts[((long)b * CIN + c) * NN + idx];
    float ce = newpts[(long)bs * CIN + c];
    _Float16* Ar = A + row * 160;
    Ar[c]      = (_Float16)(da * (pv - ce));
    Ar[64 + c] = (_Float16)(ca * ce);
    if (c < 32) {
        float g = 0.f;
        if (c < 3)
            g = (xyz[((long)b * 3 + c) * NN + idx] - newxyz[(long)bs * 3 + c]) * inv_radius;
        Ar[128 + c] = (_Float16)g;
    }
}

// ---------------------------------------------------------------------------
// 4. Pack weights [Cout,Cin] f32 -> [Cout,Cinpad] f16, zero padded columns.
// ---------------------------------------------------------------------------
__global__ void pack_w_kernel(const float* __restrict__ W, int Cin, int Cinpad,
                              int total, _Float16* __restrict__ Wp)
{
    int t = blockIdx.x * 256 + threadIdx.x;
    if (t >= total) return;
    int n = t / Cinpad;
    int k = t % Cinpad;
    Wp[t] = (k < Cin) ? (_Float16)W[(long)n * Cin + k] : (_Float16)0.f;
}

// ---------------------------------------------------------------------------
// 5. GEMM via V_WMMA_F32_16X16X32_F16.
//    X[M,Cout] = A[M,Cinpad](f16) * Wp[Cout,Cinpad](f16)^T   (no bias: it
//    cancels inside train-mode BN).
//    Each wave computes a 16-row strip across the FULL output width:
//    one A-fragment load feeds NT (= Cout/16) WMMAs, maximizing the
//    wmma:vmem ratio. Grid: (M/128); 8 waves/block.
// ---------------------------------------------------------------------------
template<int NT>
__global__ void gemm_wmma_kernel(const _Float16* __restrict__ A, int Cinpad,
                                 const _Float16* __restrict__ Wp,
                                 float* __restrict__ X)
{
    const int Cout = NT * 16;
    int lane = threadIdx.x & 31;
    int wave = threadIdx.x >> 5;
    long mTile = (long)blockIdx.x * 8 + wave;
    int  ml   = lane & 15;      // row-within-tile for A, col-within-tile for B
    int  half = lane >> 4;

    // Per-lane base pointers (dword = 2 consecutive f16 along K; K offsets even
    // and Cinpad % 32 == 0 so all dword accesses are 4B aligned).
    const uint* Arow = (const uint*)(A + (mTile * 16 + ml) * (long)Cinpad);

    v8f acc[NT] = {};
    for (int k0 = 0; k0 < Cinpad; k0 += 32) {
        int kb = k0 >> 1;                 // dword index of k0
        v16h a;
        #pragma unroll
        for (int v = 0; v < 8; ++v) {
            // A 16x32 f16 layout: VGPR v<4 -> K = 2v,2v+1 (+8*half);
            //                     VGPR v>=4 -> K = 16+2(v-4),.. (+8*half)
            int ai = kb + ((v < 4) ? v : (4 + v)) + 4 * half;
            union { uint u; _Float16 h[2]; } ua;
            ua.u = Arow[ai];
            a[2 * v]     = ua.h[0];
            a[2 * v + 1] = ua.h[1];
        }
        // prefetch next K-block of this A row (global_prefetch_b8)
        if (k0 + 32 < Cinpad)
            __builtin_prefetch((const void*)(Arow + kb + 16), 0, 3);

        #pragma unroll
        for (int nt = 0; nt < NT; ++nt) {
            const uint* Brow = (const uint*)(Wp + ((long)nt * 16 + ml) * (long)Cinpad);
            v16h bf;
            #pragma unroll
            for (int v = 0; v < 8; ++v) {
                // B 32x16 f16 layout (same as A): VGPR v<4 -> K = 2v,2v+1 (+8*half);
                // VGPR v>=4 -> K = 16+2(v-4),.. (+8*half)
                union { uint u; _Float16 h[2]; } ub;
                ub.u = Brow[kb + ((v < 4) ? v : (4 + v)) + 4 * half];
                bf[2 * v]     = ub.h[0];
                bf[2 * v + 1] = ub.h[1];
            }
            acc[nt] = __builtin_amdgcn_wmma_f32_16x16x32_f16(
                          false, a, false, bf, (short)0, acc[nt], false, false);
        }
    }
    // C/D f32 layout: VGPR r, lanes 0-15 -> M=r, lanes 16-31 -> M=8+r; N=lane%16
    #pragma unroll
    for (int nt = 0; nt < NT; ++nt) {
        #pragma unroll
        for (int r = 0; r < 8; ++r)
            X[(mTile * 16 + r + 8 * half) * (long)Cout + nt * 16 + ml] = acc[nt][r];
    }
}

// ---------------------------------------------------------------------------
// 6a. Deterministic per-channel partial sums (fixed block count, fixed order).
//     blockDim = (Cout, 4); grid = 1024; rowsPerBlock = M/1024.
// ---------------------------------------------------------------------------
__global__ void stats_partial_kernel(const float* __restrict__ X, int Cout,
                                     int rowsPerBlock, float* __restrict__ partial)
{
    __shared__ float sS[4 * 128];
    __shared__ float sQ[4 * 128];
    int c  = threadIdx.x;
    int ty = threadIdx.y;
    long r0 = (long)blockIdx.x * rowsPerBlock;
    float s = 0.f, q = 0.f;
    for (int j = ty; j < rowsPerBlock; j += 4) {
        float x = X[(r0 + j) * Cout + c];
        s += x;
        q += x * x;
    }
    sS[ty * Cout + c] = s;
    sQ[ty * Cout + c] = q;
    __syncthreads();
    if (ty == 0) {
        float ss = sS[c] + sS[Cout + c] + sS[2 * Cout + c] + sS[3 * Cout + c];
        float qq = sQ[c] + sQ[Cout + c] + sQ[2 * Cout + c] + sQ[3 * Cout + c];
        partial[(long)blockIdx.x * 2 * Cout + c]        = ss;
        partial[(long)blockIdx.x * 2 * Cout + Cout + c] = qq;
    }
}

// 6b. Serial (deterministic) finalize: accum[c] = sum, accum[Cout+c] = sumsq.
__global__ void stats_final_kernel(const float* __restrict__ partial, int Cout,
                                   float* __restrict__ accum)
{
    int c = threadIdx.x;
    if (c >= Cout) return;
    float s = 0.f, q = 0.f;
    for (int blk = 0; blk < 1024; ++blk) {
        s += partial[(long)blk * 2 * Cout + c];
        q += partial[(long)blk * 2 * Cout + Cout + c];
    }
    accum[c] = s;
    accum[Cout + c] = q;
}

// ---------------------------------------------------------------------------
// 7. BN (train-mode, population var) + ReLU, f32 -> f16 into next A buffer.
// ---------------------------------------------------------------------------
__global__ void bn_apply_kernel(const float* __restrict__ X, long total, int Cout,
                                const float* __restrict__ accum,
                                const float* __restrict__ g,
                                const float* __restrict__ bt,
                                float invN, _Float16* __restrict__ A)
{
    long i = (long)blockIdx.x * 256 + threadIdx.x;
    if (i >= total) return;
    int c = (int)(i % Cout);
    float mean = accum[c] * invN;
    float var  = accum[Cout + c] * invN - mean * mean;
    float x = X[i];
    float y = g[c] * (x - mean) * rsqrtf(var + 1e-5f) + bt[c];
    A[i] = (_Float16)fmaxf(y, 0.f);
}

// ---------------------------------------------------------------------------
// 8. Max over K, scatter into concatenated [B, 320, S] output region.
// ---------------------------------------------------------------------------
__global__ void max_k_kernel(const _Float16* __restrict__ A, int K, int Cout,
                             int coff, float* __restrict__ out)
{
    int t = blockIdx.x * 256 + threadIdx.x;
    int total = BB * SS * Cout;
    if (t >= total) return;
    int c  = t % Cout;
    int bs = t / Cout;
    int b  = bs >> 10;
    int s  = bs & 1023;
    const _Float16* p = A + (long)bs * K * Cout + c;
    float m = -3.4e38f;
    for (int k = 0; k < K; ++k) {
        float v = (float)p[(long)k * Cout];
        m = fmaxf(m, v);
    }
    out[OUT_XYZ_ELEMS + ((long)b * TOTAL_OUT_CH + coff + c) * SS + s] = m;
}

// ---------------------------------------------------------------------------
// Host launcher
// ---------------------------------------------------------------------------
extern "C" void kernel_launch(void* const* d_in, const int* in_sizes, int n_in,
                              void* d_out, int out_size, void* d_ws, size_t ws_size,
                              hipStream_t stream)
{
    (void)in_sizes; (void)n_in; (void)out_size; (void)ws_size;

    const float* xyz  = (const float*)d_in[0];   // [B,3,N]
    const float* pts  = (const float*)d_in[1];   // [B,CIN,N]
    const int*   fps  = (const int*)d_in[2];     // [B,S]
    // pytree leaf order: conv_w (9), conv_b (9, unused: cancels in BN),
    // bn_g (9), bn_b (9), diff_logit, center_logit
    const float* dlogit = (const float*)d_in[39];
    const float* clogit = (const float*)d_in[40];
    float* out = (float*)d_out;

    const float radii[3]   = {0.1f, 0.2f, 0.4f};
    const int   Ks[3]      = {16, 32, 64};
    const int   mlps[3][3] = {{32, 32, 64}, {64, 64, 128}, {64, 96, 128}};
    const int   coffs[3]   = {0, 64, 192};

    // workspace carve-up (256B aligned)
    size_t off = 0;
    char* ws = (char*)d_ws;
    auto carve = [&](size_t bytes) -> void* {
        void* p = ws + off;
        off += (bytes + 255) & ~(size_t)255;
        return p;
    };
    float*     newxyz  = (float*)carve((size_t)BB * SS * 3 * 4);
    float*     newpts  = (float*)carve((size_t)BB * SS * CIN * 4);
    int*       gidx    = (int*)carve((size_t)BB * SS * 64 * 4);
    _Float16*  Wp      = (_Float16*)carve((size_t)128 * 160 * 2);
    float*     partial = (float*)carve((size_t)1024 * 2 * 128 * 4);
    float*     accum   = (float*)carve((size_t)256 * 4);
    _Float16*  bufA    = (_Float16*)carve((size_t)BB * SS * 64 * 160 * 2);
    float*     bufX    = (float*)carve((size_t)BB * SS * 64 * 128 * 4);

    // 1) gather
    gather_new_kernel<<<(BB * SS + 255) / 256, 256, 0, stream>>>(
        xyz, pts, fps, newxyz, newpts, out);

    for (int i = 0; i < 3; ++i) {
        const int K = Ks[i];
        const long M = (long)BB * SS * K;          // 131072 / 262144 / 524288
        const float r = radii[i];

        // 2) ball query: one wave per query, 8 waves per block
        ball_query_kernel<<<(BB * SS) / 8, 256, 0, stream>>>(
            xyz, newxyz, gidx, r * r, K);

        // 3) build features [M x 160] f16
        {
            dim3 blk(64, 4);
            feat_build_kernel<<<(unsigned)(M / 4), blk, 0, stream>>>(
                xyz, pts, newxyz, newpts, gidx, dlogit, clogit, 1.0f / r, K, bufA);
        }

        // 4-7) MLP layers: GEMM (WMMA) -> BN stats -> BN+ReLU
        int Cin_l = 131, Cinpad_l = 160;
        for (int l = 0; l < 3; ++l) {
            const int Cout = mlps[i][l];
            const float* W  = (const float*)d_in[3  + i * 3 + l];
            const float* g  = (const float*)d_in[21 + i * 3 + l];
            const float* bt = (const float*)d_in[30 + i * 3 + l];

            int wtotal = Cout * Cinpad_l;
            pack_w_kernel<<<(wtotal + 255) / 256, 256, 0, stream>>>(
                W, Cin_l, Cinpad_l, wtotal, Wp);

            {
                dim3 grid((unsigned)(M / 128));
                switch (Cout / 16) {
                case 2:
                    gemm_wmma_kernel<2><<<grid, 256, 0, stream>>>(bufA, Cinpad_l, Wp, bufX);
                    break;
                case 4:
                    gemm_wmma_kernel<4><<<grid, 256, 0, stream>>>(bufA, Cinpad_l, Wp, bufX);
                    break;
                case 6:
                    gemm_wmma_kernel<6><<<grid, 256, 0, stream>>>(bufA, Cinpad_l, Wp, bufX);
                    break;
                default:
                    gemm_wmma_kernel<8><<<grid, 256, 0, stream>>>(bufA, Cinpad_l, Wp, bufX);
                    break;
                }
            }
            {
                dim3 blk((unsigned)Cout, 4);
                stats_partial_kernel<<<1024, blk, 0, stream>>>(
                    bufX, Cout, (int)(M / 1024), partial);
            }
            stats_final_kernel<<<1, 128, 0, stream>>>(partial, Cout, accum);

            long total = M * Cout;
            bn_apply_kernel<<<(unsigned)((total + 255) / 256), 256, 0, stream>>>(
                bufX, total, Cout, accum, g, bt, 1.0f / (float)M, bufA);

            Cin_l = Cout;
            Cinpad_l = Cout;   // 32/64/96/128: all multiples of 32
        }

        // 8) max over K -> output channels [coff, coff+Cout_last)
        {
            const int CoutL = mlps[i][2];
            int total = BB * SS * CoutL;
            max_k_kernel<<<(total + 255) / 256, 256, 0, stream>>>(
                bufA, K, CoutL, coffs[i], out);
        }
    }
}
